// LSTMModel_57612691308691
// MI455X (gfx1250) — matrix operand order repaired and checked by reading, not compiled, not run on hardware
//
#include <hip/hip_runtime.h>
#include <hip/hip_bf16.h>

typedef __attribute__((ext_vector_type(16))) _Float16 v16h;
typedef __attribute__((ext_vector_type(8)))  float    v8f;

#define B_   64
#define T_   1024
#define H_   256
#define G4_  1024

static __device__ __forceinline__ float sigmoidf_(float x) {
    return 1.0f / (1.0f + __expf(-x));
}
static __device__ __forceinline__ float eluf_(float x) {
    return x > 0.0f ? x : (__expf(x) - 1.0f);
}

union AFrag { uint4 u[2]; v16h h; };

// Load the 8 B fragments (2 jt-combos x 4 gates) of k-chunk kc for this lane.
// Fragments are pre-swizzled so each lane's 16 f16 are contiguous (32B).
static __device__ __forceinline__ void load_bfrags(const _Float16* __restrict__ W,
                                                   int kc, int jb, int lane,
                                                   v16h* __restrict__ dst) {
#pragma unroll
    for (int ci = 0; ci < 2; ++ci)
#pragma unroll
        for (int g = 0; g < 4; ++g) {
            int nt = g * 16 + jb + 8 * ci;
            dst[ci * 4 + g] =
                *(const v16h*)&W[(size_t)(((kc * 64 + nt) * 32) + lane) * 16];
        }
}

static __device__ __forceinline__ v16h load_afrag(const _Float16* __restrict__ abuf,
                                                  int m, int ln, int half, int kcl) {
    AFrag af;
    int abase = (m * 16 + ln) * H_ + kcl * 32 + 16 * half;
    af.u[0] = *(const uint4*)&abuf[abase];
    af.u[1] = *(const uint4*)&abuf[abase + 8];
    return af.h;
}

// ---------------------------------------------------------------------------
// Swizzle a [kcs*32, 1024] f32 row-major weight matrix into WMMA B-fragment
// order (f16): dst[((kc*64+nt)*32+lane)*16+j] = W[(kc*32+j+16*(lane>>4))*1024
//                                                 + nt*16 + (lane&15)]
// ---------------------------------------------------------------------------
__global__ void swizzle_w(const float* __restrict__ src, _Float16* __restrict__ dst,
                          int kcs) {
    int e = blockIdx.x * blockDim.x + threadIdx.x;
    int total = kcs * 64 * 32 * 16;
    if (e >= total) return;
    int j    = e & 15;
    int lane = (e >> 4) & 31;
    int nt   = (e >> 9) & 63;
    int kc   = e >> 15;
    int K = kc * 32 + j + 16 * (lane >> 4);
    int N = nt * 16 + (lane & 15);
    dst[e] = (_Float16)src[K * 1024 + N];
}

// ---------------------------------------------------------------------------
// LSTM layer 0 (input dim 1). Persistent single workgroup, 32 waves.
// Wave w: m-tile = w&3, h-column tiles jt in {w>>2, (w>>2)+8}; each combo
// computes its i/f/g/o tiles so the cell update is wave-local (c in regs).
// ---------------------------------------------------------------------------
__global__ void __launch_bounds__(1024) lstm0_kernel(
    const float* __restrict__ x,        // [64,1024]
    const _Float16* __restrict__ WhF,   // swizzled Wh0, 8 k-chunks
    const float* __restrict__ Wx0,      // [1024]
    const float* __restrict__ b0,       // [1024]
    _Float16* __restrict__ h0seq)       // [T,64,256] f16 out
{
    __shared__ _Float16 hbuf[B_ * H_];

    const int tid  = threadIdx.x;
    const int wave = tid >> 5, lane = tid & 31;
    const int m    = wave & 3;
    const int jb   = wave >> 2;             // 0..7
    const int half = lane >> 4, ln = lane & 15;

    for (int i = tid; i < B_ * H_; i += 1024) hbuf[i] = (_Float16)0.0f;

    float wgt[2][4], bia[2][4];
    v8f cst[2], z8 = {};
#pragma unroll
    for (int ci = 0; ci < 2; ++ci) {
        int col = (jb + 8 * ci) * 16 + ln;
#pragma unroll
        for (int g = 0; g < 4; ++g) {
            wgt[ci][g] = Wx0[g * 256 + col];
            bia[ci][g] = b0 [g * 256 + col];
        }
        cst[ci] = z8;
    }
    __syncthreads();

    for (int t = 0; t < T_; ++t) {
        // Opaque copy of the weight pointer: stops LICM from hoisting the
        // (t-invariant) B-fragment loads out of the time loop and spilling.
        const _Float16* Wt = WhF;
        asm volatile("" : "+s"(Wt));

        v8f acc[2][4];
#pragma unroll
        for (int ci = 0; ci < 2; ++ci)
#pragma unroll
            for (int g = 0; g < 4; ++g) acc[ci][g] = z8;

        for (int kc = 0; kc < 8; ++kc) {
            v16h bf[8];
            load_bfrags(Wt, kc, jb, lane, bf);
            v16h a = load_afrag(hbuf, m, ln, half, kc);
#pragma unroll
            for (int ci = 0; ci < 2; ++ci)
#pragma unroll
                for (int g = 0; g < 4; ++g)
                    acc[ci][g] = __builtin_amdgcn_wmma_f32_16x16x32_f16(
                        false, a, false, bf[ci * 4 + g], (short)0,
                        acc[ci][g], false, false);
        }
        __syncthreads();

#pragma unroll
        for (int ci = 0; ci < 2; ++ci) {
            int col = (jb + 8 * ci) * 16 + ln;
            v8f c = cst[ci];
#pragma unroll
            for (int r = 0; r < 8; ++r) {
                int brow = m * 16 + r + 8 * half;
                float xv = x[brow * T_ + t];
                float pi = acc[ci][0][r] + xv * wgt[ci][0] + bia[ci][0];
                float pf = acc[ci][1][r] + xv * wgt[ci][1] + bia[ci][1];
                float pg = acc[ci][2][r] + xv * wgt[ci][2] + bia[ci][2];
                float po = acc[ci][3][r] + xv * wgt[ci][3] + bia[ci][3];
                float ig = sigmoidf_(pi), fg = sigmoidf_(pf);
                float gg = tanhf(pg),     og = sigmoidf_(po);
                float cn = fg * c[r] + ig * gg;
                c[r] = cn;
                _Float16 hh = (_Float16)(og * tanhf(cn));
                hbuf[brow * H_ + col] = hh;
                h0seq[(size_t)t * (B_ * H_) + brow * H_ + col] = hh;
            }
            cst[ci] = c;
        }
        __syncthreads();
    }
}

// ---------------------------------------------------------------------------
// LSTM layer 1: fused K = [h0_t | h1_prev] (512) against stacked [Wx1; Wh1]
// (16 swizzled k-chunks). Time-mean of h1 accumulates in registers.
// ---------------------------------------------------------------------------
__global__ void __launch_bounds__(1024) lstm1_kernel(
    const _Float16* __restrict__ W1F,   // 16 k-chunk fragments ([Wx1;Wh1])
    const float* __restrict__ b1,       // [1024]
    const _Float16* __restrict__ h0seq, // [T,64,256] f16
    float* __restrict__ hmean)          // [64,256] f32 out
{
    __shared__ _Float16 h0buf[B_ * H_];
    __shared__ _Float16 hbuf [B_ * H_];

    const int tid  = threadIdx.x;
    const int wave = tid >> 5, lane = tid & 31;
    const int m    = wave & 3;
    const int jb   = wave >> 2;
    const int half = lane >> 4, ln = lane & 15;

    for (int i = tid; i < B_ * H_; i += 1024) hbuf[i] = (_Float16)0.0f;

    float bia[2][4];
    v8f cst[2], macc[2], z8 = {};
#pragma unroll
    for (int ci = 0; ci < 2; ++ci) {
        int col = (jb + 8 * ci) * 16 + ln;
#pragma unroll
        for (int g = 0; g < 4; ++g) bia[ci][g] = b1[g * 256 + col];
        cst[ci] = z8;  macc[ci] = z8;
    }

    for (int t = 0; t < T_; ++t) {
        const _Float16* Wt = W1F;
        asm volatile("" : "+s"(Wt));

        // cooperative load of h0_t into LDS (32KB -> 2 x uint4 per thread)
        const uint4* srcp = (const uint4*)&h0seq[(size_t)t * (B_ * H_)];
        uint4* dstp = (uint4*)h0buf;
        dstp[tid]        = srcp[tid];
        dstp[tid + 1024] = srcp[tid + 1024];
        if (t + 1 < T_)
            __builtin_prefetch(&h0seq[(size_t)(t + 1) * (B_ * H_) + tid * 8], 0, 1);
        __syncthreads();

        v8f acc[2][4];
#pragma unroll
        for (int ci = 0; ci < 2; ++ci)
#pragma unroll
            for (int g = 0; g < 4; ++g) acc[ci][g] = z8;

        for (int kc = 0; kc < 16; ++kc) {
            v16h bf[8];
            load_bfrags(Wt, kc, jb, lane, bf);
            const _Float16* abuf = (kc < 8) ? h0buf : hbuf;
            v16h a = load_afrag(abuf, m, ln, half, kc & 7);
#pragma unroll
            for (int ci = 0; ci < 2; ++ci)
#pragma unroll
                for (int g = 0; g < 4; ++g)
                    acc[ci][g] = __builtin_amdgcn_wmma_f32_16x16x32_f16(
                        false, a, false, bf[ci * 4 + g], (short)0,
                        acc[ci][g], false, false);
        }
        __syncthreads();

#pragma unroll
        for (int ci = 0; ci < 2; ++ci) {
            int col = (jb + 8 * ci) * 16 + ln;
            v8f c = cst[ci], mv = macc[ci];
#pragma unroll
            for (int r = 0; r < 8; ++r) {
                int brow = m * 16 + r + 8 * half;
                float pi = acc[ci][0][r] + bia[ci][0];
                float pf = acc[ci][1][r] + bia[ci][1];
                float pg = acc[ci][2][r] + bia[ci][2];
                float po = acc[ci][3][r] + bia[ci][3];
                float ig = sigmoidf_(pi), fg = sigmoidf_(pf);
                float gg = tanhf(pg),     og = sigmoidf_(po);
                float cn = fg * c[r] + ig * gg;
                c[r] = cn;
                float hv = og * tanhf(cn);
                mv[r] += hv;
                hbuf[brow * H_ + col] = (_Float16)hv;
            }
            cst[ci] = c;  macc[ci] = mv;
        }
        __syncthreads();
    }

#pragma unroll
    for (int ci = 0; ci < 2; ++ci) {
        int col = (jb + 8 * ci) * 16 + ln;
#pragma unroll
        for (int r = 0; r < 8; ++r) {
            int brow = m * 16 + r + 8 * half;
            hmean[brow * H_ + col] = macc[ci][r] * (1.0f / (float)T_);
        }
    }
}

// ---------------------------------------------------------------------------
// Head: tproj -> concat -> 3x (dense + elu) -> dense. ~30 MFLOP, plain f32.
// ---------------------------------------------------------------------------
__global__ void __launch_bounds__(256) head_kernel(
    const float* __restrict__ hmean,   // [64,256]
    const float* __restrict__ theta,   // [64,5]
    const float* __restrict__ projW, const float* __restrict__ projb,
    const float* __restrict__ l0W,   const float* __restrict__ l0b,
    const float* __restrict__ l1W,   const float* __restrict__ l1b,
    const float* __restrict__ l2W,   const float* __restrict__ l2b,
    const float* __restrict__ oW,    const float* __restrict__ ob,
    float* __restrict__ out)           // [64]
{
    __shared__ float bufA[64 * 256];
    __shared__ float bufB[64 * 256];
    const int tid = threadIdx.x;

    for (int i = tid; i < 64 * 256; i += 256) {
        int b = i >> 8, j = i & 255;
        float s = projb[j];
#pragma unroll
        for (int k = 0; k < 5; ++k) s += theta[b * 5 + k] * projW[k * 256 + j];
        bufA[i] = s;
    }
    __syncthreads();

    for (int i = tid; i < 64 * 256; i += 256) {
        int b = i >> 8, j = i & 255;
        float s = l0b[j];
        for (int k = 0; k < 256; ++k) s += hmean[b * 256 + k] * l0W[k * 256 + j];
        for (int k = 0; k < 256; ++k) s += bufA[b * 256 + k] * l0W[(256 + k) * 256 + j];
        bufB[i] = eluf_(s);
    }
    __syncthreads();

    for (int i = tid; i < 64 * 256; i += 256) {
        int b = i >> 8, j = i & 255;
        float s = l1b[j];
        for (int k = 0; k < 256; ++k) s += bufB[b * 256 + k] * l1W[k * 256 + j];
        bufA[i] = eluf_(s);
    }
    __syncthreads();

    for (int i = tid; i < 64 * 128; i += 256) {
        int b = i >> 7, j = i & 127;
        float s = l2b[j];
        for (int k = 0; k < 256; ++k) s += bufA[b * 256 + k] * l2W[k * 128 + j];
        bufB[i] = eluf_(s);
    }
    __syncthreads();

    for (int b = tid; b < 64; b += 256) {
        float s = ob[0];
        for (int k = 0; k < 128; ++k) s += bufB[b * 128 + k] * oW[k];
        out[b] = s;
    }
}

// ---------------------------------------------------------------------------
extern "C" void kernel_launch(void* const* d_in, const int* in_sizes, int n_in,
                              void* d_out, int out_size, void* d_ws, size_t ws_size,
                              hipStream_t stream) {
    const float* x     = (const float*)d_in[0];
    const float* theta = (const float*)d_in[1];
    const float* l0Wx  = (const float*)d_in[2];
    const float* l0Wh  = (const float*)d_in[3];
    const float* l0b   = (const float*)d_in[4];
    const float* l1Wx  = (const float*)d_in[5];
    const float* l1Wh  = (const float*)d_in[6];
    const float* l1b   = (const float*)d_in[7];
    const float* projW = (const float*)d_in[8];
    const float* projb = (const float*)d_in[9];
    const float* lin0W = (const float*)d_in[10];
    const float* lin0b = (const float*)d_in[11];
    const float* lin1W = (const float*)d_in[12];
    const float* lin1b = (const float*)d_in[13];
    const float* lin2W = (const float*)d_in[14];
    const float* lin2b = (const float*)d_in[15];
    const float* outW  = (const float*)d_in[16];
    const float* outb  = (const float*)d_in[17];

    char* ws = (char*)d_ws;
    _Float16* WhF0  = (_Float16*)ws;                              // 512 KB
    _Float16* W1F   = (_Float16*)(ws + (512u << 10));             // 1 MB
    _Float16* h0seq = (_Float16*)(ws + (1536u << 10));            // 32 MB
    float*    hmean = (float*)(ws + (1536u << 10) + (32u << 20)); // 64 KB

    const int thr = 256;
    const int fragElems = 8 * 64 * 32 * 16;   // 262144 per 8 k-chunks
    swizzle_w<<<(fragElems + thr - 1) / thr, thr, 0, stream>>>(l0Wh, WhF0, 8);
    swizzle_w<<<(fragElems + thr - 1) / thr, thr, 0, stream>>>(l1Wx, W1F, 8);
    swizzle_w<<<(fragElems + thr - 1) / thr, thr, 0, stream>>>(l1Wh, W1F + fragElems, 8);

    lstm0_kernel<<<1, 1024, 0, stream>>>(x, WhF0, l0Wx, l0b, h0seq);
    lstm1_kernel<<<1, 1024, 0, stream>>>(W1F, l1b, h0seq, hmean);
    head_kernel<<<1, 256, 0, stream>>>(hmean, theta, projW, projb,
                                       lin0W, lin0b, lin1W, lin1b,
                                       lin2W, lin2b, outW, outb,
                                       (float*)d_out);
}
